// Waveletnet_75153337745941
// MI455X (gfx1250) — compile-verified, matrix operand order checked by reading, not run on hardware
//
#include <hip/hip_runtime.h>

typedef __attribute__((ext_vector_type(16))) _Float16 v16h;
typedef __attribute__((ext_vector_type(8)))  _Float16 v8h;
typedef __attribute__((ext_vector_type(8)))  float    v8f;
typedef __attribute__((ext_vector_type(4)))  float    v4f;
typedef int v4i_gcc __attribute__((vector_size(16)));   // matches async-LDS builtin pointee

#if defined(__gfx1250__) && __has_builtin(__builtin_amdgcn_global_load_async_to_lds_b128) && \
    __has_builtin(__builtin_amdgcn_s_wait_asynccnt)
#define USE_ASYNC_LDS 1
#else
#define USE_ASYNC_LDS 0
#endif

// ====================================================================
// Weight pre-pack: fp32 OIHW (Cout,Cin,3,3) -> f16 per-lane WMMA
// A-fragment images: [tap(9)][cinChunk][coutTile][lane(32)][half(16)]
// For lane L: row M = L%16 (cout), kb = (L<16?0:8);
//   halves 0..7  -> K = kb + h, halves 8..15 -> K = kb + 16 + (h-8)
// Out-of-range cout/cin -> 0 (exact-zero padded outputs).
// ====================================================================
__global__ void pack_w_kernel(const float* __restrict__ w, _Float16* __restrict__ pk,
                              int Cout, int Cin, int cinChunks, int coutTiles, int total)
{
    int idx = blockIdx.x * blockDim.x + threadIdx.x;
    if (idx >= total) return;
    int h    = idx & 15;
    int lane = (idx >> 4) & 31;
    int rest = idx >> 9;
    int mt   = rest % coutTiles;  rest /= coutTiles;
    int kc   = rest % cinChunks;
    int tap  = rest / cinChunks;
    int cout = mt * 16 + (lane & 15);
    int kb   = (lane >> 4) << 3;
    int cin  = (kc << 5) + kb + ((h < 8) ? h : (8 + h));
    float v = 0.0f;
    if (cout < Cout && cin < Cin)
        v = w[((size_t)cout * Cin + cin) * 9 + tap];
    pk[idx] = (_Float16)v;
}

__global__ void zero_fill_kernel(float* __restrict__ p, int n)
{
    int i = blockIdx.x * blockDim.x + threadIdx.x;
    if (i < n) p[i] = 0.0f;
}

__global__ void pad_bias_kernel(const float* __restrict__ b, float* __restrict__ o,
                                int Cout, int n)
{
    int i = blockIdx.x * blockDim.x + threadIdx.x;
    if (i < n) o[i] = (i < Cout) ? b[i] : 0.0f;
}

// --------------------------------------------------------------------
// Shared conv epilogue: bias + optional residual + lrelu + f16 store.
// Lane owns cout = coutBase + hi*8 + r (r=0..7) at pixel pix.
// --------------------------------------------------------------------
__device__ __forceinline__ void conv_epilogue(
    const v8f* acc, int mt0, int hi, size_t pix,
    const float* __restrict__ biasPad, _Float16* __restrict__ out,
    const _Float16* __restrict__ addend,
    int outStride, int addStride, int doLrelu)
{
#pragma unroll
    for (int m = 0; m < 2; ++m) {
        const int coutBase = (mt0 + m) << 4;
        const float* bb = biasPad + coutBase + hi * 8;
        const v4f b0 = *(const v4f*)bb;
        const v4f b1 = *(const v4f*)(bb + 4);
        float res[8];
#pragma unroll
        for (int r = 0; r < 8; ++r)
            res[r] = acc[m][r] + ((r < 4) ? b0[r & 3] : b1[r & 3]);
        if (addend) {
            const v8h av = *(const v8h*)(addend + pix * addStride + coutBase + hi * 8);
#pragma unroll
            for (int r = 0; r < 8; ++r) res[r] += (float)av[r];
        }
        if (doLrelu) {
#pragma unroll
            for (int r = 0; r < 8; ++r) res[r] = (res[r] >= 0.0f) ? res[r] : 0.2f * res[r];
        }
        v8h o;
#pragma unroll
        for (int r = 0; r < 8; ++r) o[r] = (_Float16)res[r];
        *(v8h*)(out + pix * outStride + coutBase + hi * 8) = o;
    }
}

// ====================================================================
// Direct implicit-GEMM 3x3 conv (small layers).  One wave == 2 M-tiles
// x 16 x-positions; 4 waves/block on adjacent spatial tiles.  Halo via
// zero-page pointer select (branch-free, unconditional loads).
// ====================================================================
__global__ __launch_bounds__(128) void conv3x3_wmma_kernel(
    const _Float16* __restrict__ in, const _Float16* __restrict__ wpack,
    const float* __restrict__ biasPad, _Float16* __restrict__ out,
    const _Float16* __restrict__ addend, const _Float16* __restrict__ zpage,
    int H, int W, int inStride, int cinChunks,
    int outStride, int addStride, int doLrelu)
{
    const int lane   = threadIdx.x & 31;
    const int waveId = threadIdx.x >> 5;
    const int n      = lane & 15;
    const int hi     = lane >> 4;
    const int kb     = hi << 3;
    const int wTiles = W >> 4;

    int nt = blockIdx.x * 4 + waveId;
    const int x = ((nt % wTiles) << 4) + n;
    nt /= wTiles;
    const int y = nt % H;
    const int b = nt / H;

    const int mt0       = blockIdx.y << 1;
    const int coutTiles = gridDim.y << 1;

    const _Float16* bptr[9];
#pragma unroll
    for (int tap = 0; tap < 9; ++tap) {
        const int ky = tap / 3 - 1;
        const int kx = tap % 3 - 1;
        const int gy = y + ky, gx = x + kx;
        const bool ok = ((unsigned)gy < (unsigned)H) & ((unsigned)gx < (unsigned)W);
        bptr[tap] = ok ? (in + ((size_t)(b * H + gy) * W + gx) * inStride + kb)
                       : (zpage + kb);
    }

    v8f acc[2];
    acc[0] = (v8f){0.f, 0.f, 0.f, 0.f, 0.f, 0.f, 0.f, 0.f};
    acc[1] = (v8f){0.f, 0.f, 0.f, 0.f, 0.f, 0.f, 0.f, 0.f};

    const _Float16* __restrict__ aLane = wpack + ((size_t)mt0 * 32 + lane) * 16;
    const size_t aKcStep  = (size_t)coutTiles * 512;
    const size_t aTapStep = aKcStep * (size_t)cinChunks;

    for (int kc = 0; kc < cinChunks; ++kc) {
        const int cb = kc << 5;
        const _Float16* aKc = aLane + (size_t)kc * aKcStep;
        __builtin_prefetch(aKc + aKcStep, 0, 3);
#pragma unroll
        for (int tap = 0; tap < 9; ++tap) {
            const _Float16* ap = aKc + (size_t)tap * aTapStep;
            const v16h A0 = *(const v16h*)ap;
            const v16h A1 = *(const v16h*)(ap + 512);
            union { v16h v; v8h h[2]; } Bf;
            const _Float16* bp = bptr[tap] + cb;
            Bf.h[0] = *(const v8h*)bp;
            Bf.h[1] = *(const v8h*)(bp + 16);
            acc[0] = __builtin_amdgcn_wmma_f32_16x16x32_f16(
                false, A0, false, Bf.v, (short)0, acc[0], false, false);
            acc[1] = __builtin_amdgcn_wmma_f32_16x16x32_f16(
                false, A1, false, Bf.v, (short)0, acc[1], false, false);
        }
    }

    const size_t pix = ((size_t)b * H + y) * W + x;
    conv_epilogue(acc, mt0, hi, pix, biasPad, out, addend, outStride, addStride, doLrelu);
}

// ====================================================================
// LDS-staged implicit-GEMM 3x3 conv (heavy layers, coutTiles >= 8).
// Block = 4 waves sharing ONE spatial tile, covering 8 cout tiles.
// Per cin-chunk the 3x18x32ch B halo slab (3456B) is staged once into
// a double-buffered LDS region — async (ASYNCcnt) when available —
// then consumed 9 taps x 8 M-tiles from ds_load_b128.
// ====================================================================
__global__ __launch_bounds__(128) void conv3x3_wmma_lds_kernel(
    const _Float16* __restrict__ in, const _Float16* __restrict__ wpack,
    const float* __restrict__ biasPad, _Float16* __restrict__ out,
    const _Float16* __restrict__ addend, const _Float16* __restrict__ zpage,
    int H, int W, int inStride, int cinChunks,
    int outStride, int addStride, int doLrelu)
{
    __shared__ _Float16 smem[2][3 * 18 * 32];

    const int tid  = threadIdx.x;
    const int lane = tid & 31;
    const int wave = tid >> 5;
    const int n    = lane & 15;
    const int hi   = lane >> 4;
    const int kb   = hi << 3;
    const int wTiles = W >> 4;

    int nt = blockIdx.x;
    const int xBase = (nt % wTiles) << 4;
    nt /= wTiles;
    const int y = nt % H;
    const int b = nt / H;

    const int mt0       = (blockIdx.y << 3) + (wave << 1);  // 8 tiles/block, 2/wave
    const int coutTiles = gridDim.y << 3;

    // ---- staging task setup: 54 halo pixels x 4 x 16B segments = 216 tasks
    const _Float16* g0; int l0;
    const _Float16* g1; int l1;
    {
        const int t0 = tid;                 // always < 216
        const int seg = t0 & 3, pix = t0 >> 2;
        const int r = pix / 18, c = pix % 18;
        const int gy = y + r - 1, gx = xBase + c - 1;
        const bool ok = ((unsigned)gy < (unsigned)H) & ((unsigned)gx < (unsigned)W);
        g0 = ok ? (in + ((size_t)(b * H + gy) * W + gx) * inStride + seg * 8)
                : (zpage + seg * 8);
        l0 = (r * 18 + c) * 32 + seg * 8;
    }
    {
        const int t1 = tid + 128;           // valid iff tid < 88
        const int seg = t1 & 3, pix = t1 >> 2;
        const int r = pix / 18, c = pix % 18;
        const int gy = y + r - 1, gx = xBase + c - 1;
        const bool ok = ((unsigned)gy < (unsigned)H) & ((unsigned)gx < (unsigned)W);
        g1 = ok ? (in + ((size_t)(b * H + gy) * W + gx) * inStride + seg * 8)
                : (zpage + seg * 8);
        l1 = (pix < 54 ? (r * 18 + c) * 32 + seg * 8 : 0);
    }

    v8f acc[2];
    acc[0] = (v8f){0.f, 0.f, 0.f, 0.f, 0.f, 0.f, 0.f, 0.f};
    acc[1] = (v8f){0.f, 0.f, 0.f, 0.f, 0.f, 0.f, 0.f, 0.f};

    const _Float16* __restrict__ aLane = wpack + ((size_t)mt0 * 32 + lane) * 16;
    const size_t aKcStep  = (size_t)coutTiles * 512;
    const size_t aTapStep = aKcStep * (size_t)cinChunks;

    auto stage = [&](int kc, int bufIdx) {
        const int cb = kc << 5;
#if USE_ASYNC_LDS
        __builtin_amdgcn_global_load_async_to_lds_b128(
            (__attribute__((address_space(1))) v4i_gcc*)(g0 + cb),
            (__attribute__((address_space(3))) v4i_gcc*)(&smem[bufIdx][l0]), 0, 0);
        if (tid < 88)
            __builtin_amdgcn_global_load_async_to_lds_b128(
                (__attribute__((address_space(1))) v4i_gcc*)(g1 + cb),
                (__attribute__((address_space(3))) v4i_gcc*)(&smem[bufIdx][l1]), 0, 0);
#else
        *(v8h*)(&smem[bufIdx][l0]) = *(const v8h*)(g0 + cb);
        if (tid < 88)
            *(v8h*)(&smem[bufIdx][l1]) = *(const v8h*)(g1 + cb);
#endif
    };
    auto compute = [&](int kc, int bufIdx) {
        const _Float16* aKc = aLane + (size_t)kc * aKcStep;
        const _Float16* sb = &smem[bufIdx][0];
#pragma unroll
        for (int tap = 0; tap < 9; ++tap) {
            const int row = tap / 3;            // 0..2 == 1+(ky-1)
            const int col = tap % 3 + n;        // 0..17 == 1+(kx-1)+n
            const _Float16* ap = aKc + (size_t)tap * aTapStep;
            const v16h A0 = *(const v16h*)ap;
            const v16h A1 = *(const v16h*)(ap + 512);
            union { v16h v; v8h h[2]; } Bf;
            const _Float16* bp = sb + (row * 18 + col) * 32 + kb;
            Bf.h[0] = *(const v8h*)bp;          // ds_load_b128
            Bf.h[1] = *(const v8h*)(bp + 16);   // ds_load_b128
            acc[0] = __builtin_amdgcn_wmma_f32_16x16x32_f16(
                false, A0, false, Bf.v, (short)0, acc[0], false, false);
            acc[1] = __builtin_amdgcn_wmma_f32_16x16x32_f16(
                false, A1, false, Bf.v, (short)0, acc[1], false, false);
        }
    };

    stage(0, 0);
#if USE_ASYNC_LDS
    __builtin_amdgcn_s_wait_asynccnt(0);
#endif
    __syncthreads();
    for (int kc = 0; kc < cinChunks; ++kc) {
        const int cur = kc & 1;
        if (kc + 1 < cinChunks) stage(kc + 1, cur ^ 1);  // prefetch next chunk
        compute(kc, cur);
#if USE_ASYNC_LDS
        __builtin_amdgcn_s_wait_asynccnt(0);
#endif
        __syncthreads();
    }

    const size_t pix = ((size_t)b * H + y) * W + (xBase + n);
    conv_epilogue(acc, mt0, hi, pix, biasPad, out, addend, outStride, addStride, doLrelu);
}

// ====================================================================
// Haar DWT on the fp32 NCHW network input -> f16 NHWC [4][128][128][32]
// ====================================================================
__global__ void wt_in_kernel(const float* __restrict__ x, _Float16* __restrict__ out)
{
    int idx = blockIdx.x * blockDim.x + threadIdx.x;
    if (idx >= 4 * 128 * 128) return;
    const int wq = idx & 127;
    const int hq = (idx >> 7) & 127;
    const int b  = idx >> 14;
    _Float16 vals[32];
#pragma unroll
    for (int i = 0; i < 32; ++i) vals[i] = (_Float16)0.0f;
    for (int c = 0; c < 3; ++c) {
        const float* px = x + (((size_t)b * 3 + c) * 256 + 2 * hq) * 256 + 2 * wq;
        const float a = px[0], bb = px[1], cc = px[256], dd = px[257];
        float k0 = 0.25f * (a + bb + cc + dd);
        float k1 = 0.5f * (a + bb - cc - dd); k1 = (k1 + 1.0f) * 0.5f;
        float k2 = 0.5f * (a - bb + cc - dd); k2 = (k2 + 1.0f) * 0.5f;
        float k3 = 0.5f * (a - bb - cc + dd); k3 = (k3 + 1.0f) * 0.5f;
        vals[4 * c + 0] = (_Float16)k0; vals[4 * c + 1] = (_Float16)k1;
        vals[4 * c + 2] = (_Float16)k2; vals[4 * c + 3] = (_Float16)k3;
    }
    _Float16* o = out + (size_t)idx * 32;
#pragma unroll
    for (int g = 0; g < 4; ++g) {
        v8h v;
#pragma unroll
        for (int e = 0; e < 8; ++e) v[e] = vals[8 * g + e];
        *(v8h*)(o + 8 * g) = v;
    }
}

// ====================================================================
// Haar DWT, f16 NHWC -> f16 NHWC (outStride == 4*C).
// ====================================================================
__global__ void wt_mid_kernel(const _Float16* __restrict__ in, _Float16* __restrict__ out,
                              int h2, int w2, int inStride, int C, int outStride, int total)
{
    int idx = blockIdx.x * blockDim.x + threadIdx.x;
    if (idx >= total) return;
    const int c = idx % C;
    int rest = idx / C;
    const int w = rest % w2; rest /= w2;
    const int h = rest % h2;
    const int b = rest / h2;
    const int Win = 2 * w2;
    const _Float16* p = in + (((size_t)(b * 2 * h2 + 2 * h)) * Win + 2 * w) * inStride + c;
    const float a  = (float)p[0];
    const float bb = (float)p[inStride];
    const float cc = (float)p[(size_t)Win * inStride];
    const float dd = (float)p[(size_t)Win * inStride + inStride];
    float k0 = 0.25f * (a + bb + cc + dd);
    float k1 = 0.5f * (a + bb - cc - dd); k1 = (k1 + 1.0f) * 0.5f;
    float k2 = 0.5f * (a - bb + cc - dd); k2 = (k2 + 1.0f) * 0.5f;
    float k3 = 0.5f * (a - bb - cc + dd); k3 = (k3 + 1.0f) * 0.5f;
    _Float16* o = out + (((size_t)(b * h2 + h)) * w2 + w) * outStride + 4 * c;
    o[0] = (_Float16)k0; o[1] = (_Float16)k1; o[2] = (_Float16)k2; o[3] = (_Float16)k3;
}

// ====================================================================
// Inverse Haar DWT into the high channel slice of a concat buffer.
// ====================================================================
__global__ void iwt_cat_kernel(const _Float16* __restrict__ in, _Float16* __restrict__ out,
                               int H, int W, int inStride, int C,
                               int outStride, int chOff, int total)
{
    int idx = blockIdx.x * blockDim.x + threadIdx.x;
    if (idx >= total) return;
    const int c = idx % C;
    int rest = idx / C;
    const int w = rest % W; rest /= W;
    const int h = rest % H;
    const int b = rest / H;
    const _Float16* p = in + (((size_t)(b * H + h)) * W + w) * inStride + 4 * c;
    const float v0 = (float)p[0];
    const float v1 = 2.0f * (float)p[1] - 1.0f;
    const float v2 = 2.0f * (float)p[2] - 1.0f;
    const float v3 = 2.0f * (float)p[3] - 1.0f;
    const float o00 = v0 + 0.5f * ( v1 + v2 + v3);
    const float o01 = v0 + 0.5f * ( v1 - v2 - v3);
    const float o10 = v0 + 0.5f * (-v1 + v2 - v3);
    const float o11 = v0 + 0.5f * (-v1 - v2 + v3);
    const int W2 = 2 * W;
    _Float16* op = out + (((size_t)(b * 2 * H + 2 * h)) * W2 + 2 * w) * outStride + chOff + c;
    op[0]                                  = (_Float16)o00;
    op[outStride]                          = (_Float16)o01;
    op[(size_t)W2 * outStride]             = (_Float16)o10;
    op[(size_t)W2 * outStride + outStride] = (_Float16)o11;
}

// ====================================================================
// Final inverse DWT (12->3 ch, 128->256) fused with 1x1 conv cfin,
// writing fp32 NCHW [4][3][256][256] to d_out.
// ====================================================================
__global__ void final_kernel(const _Float16* __restrict__ in, const float* __restrict__ cfin,
                             float* __restrict__ out)
{
    int idx = blockIdx.x * blockDim.x + threadIdx.x;
    if (idx >= 4 * 128 * 128) return;
    const int wq = idx & 127;
    const int hq = (idx >> 7) & 127;
    const int b  = idx >> 14;
    const _Float16* ip = in + (size_t)idx * 32;
    float y00[3], y01[3], y10[3], y11[3];
#pragma unroll
    for (int c = 0; c < 3; ++c) {
        const float v0 = (float)ip[4 * c + 0];
        const float v1 = 2.0f * (float)ip[4 * c + 1] - 1.0f;
        const float v2 = 2.0f * (float)ip[4 * c + 2] - 1.0f;
        const float v3 = 2.0f * (float)ip[4 * c + 3] - 1.0f;
        y00[c] = v0 + 0.5f * ( v1 + v2 + v3);
        y01[c] = v0 + 0.5f * ( v1 - v2 - v3);
        y10[c] = v0 + 0.5f * (-v1 + v2 - v3);
        y11[c] = v0 + 0.5f * (-v1 - v2 + v3);
    }
#pragma unroll
    for (int o = 0; o < 3; ++o) {
        float s00 = 0.f, s01 = 0.f, s10 = 0.f, s11 = 0.f;
#pragma unroll
        for (int c = 0; c < 3; ++c) {
            const float wv = cfin[o * 3 + c];
            s00 += wv * y00[c]; s01 += wv * y01[c];
            s10 += wv * y10[c]; s11 += wv * y11[c];
        }
        float* op = out + (((size_t)b * 3 + o) * 256 + 2 * hq) * 256 + 2 * wq;
        op[0] = s00; op[1] = s01; op[256] = s10; op[257] = s11;
    }
}

// ====================================================================
// Host-side orchestration
// ====================================================================
extern "C" void kernel_launch(void* const* d_in, const int* in_sizes, int n_in,
                              void* d_out, int out_size, void* d_ws, size_t ws_size,
                              hipStream_t stream)
{
    (void)in_sizes; (void)n_in; (void)out_size;
    const float* x    = (const float*)d_in[0];
    const float* w_c1 = (const float*)d_in[1];  const float* b_c1 = (const float*)d_in[2];
    const float* w_c2 = (const float*)d_in[3];  const float* b_c2 = (const float*)d_in[4];
    const float* w_c3 = (const float*)d_in[5];  const float* b_c3 = (const float*)d_in[6];
    const float* w_c4 = (const float*)d_in[7];  const float* b_c4 = (const float*)d_in[8];
    const float* w_d4 = (const float*)d_in[9];  const float* b_d4 = (const float*)d_in[10];
    const float* w_d3 = (const float*)d_in[11]; const float* b_d3 = (const float*)d_in[12];
    const float* w_d2 = (const float*)d_in[13]; const float* b_d2 = (const float*)d_in[14];
    const float* w_d1 = (const float*)d_in[15]; const float* b_d1 = (const float*)d_in[16];
    const float* cfin = (const float*)d_in[17];
    float* outp = (float*)d_out;

    char* base = (char*)d_ws;
    size_t used = 0;
    auto allocB = [&](size_t bytes) -> char* {
        char* r = base + used;
        used += (bytes + 255) & ~(size_t)255;
        return r;
    };
    auto allocH = [&](size_t elems) -> _Float16* { return (_Float16*)allocB(elems * 2); };
    auto allocF = [&](size_t elems) -> float*    { return (float*)allocB(elems * 4); };

    _Float16* t_w1  = allocH((size_t)4 * 128 * 128 * 32);
    _Float16* t_iw2 = allocH((size_t)4 * 128 * 128 * 32);
    _Float16* t_w2  = allocH((size_t)4 * 64 * 64 * 64);
    _Float16* t_iw3 = allocH((size_t)4 * 64 * 64 * 128);
    _Float16* t_w3  = allocH((size_t)4 * 32 * 32 * 256);
    _Float16* t_iw4 = allocH((size_t)4 * 32 * 32 * 512);
    _Float16* t_w4  = allocH((size_t)4 * 16 * 16 * 1024);
    _Float16* t_c4  = allocH((size_t)4 * 16 * 16 * 1024);
    _Float16* t_c5  = allocH((size_t)4 * 16 * 16 * 1024);
    _Float16* t_ic4 = allocH((size_t)4 * 16 * 16 * 1024);
    _Float16* t_ic3 = allocH((size_t)4 * 32 * 32 * 256);
    _Float16* t_ic2 = allocH((size_t)4 * 64 * 64 * 64);
    _Float16* t_ic1 = allocH((size_t)4 * 128 * 128 * 32);
    _Float16* t_iw1 = allocH((size_t)4 * 128 * 128 * 32);
    _Float16* p_c1 = allocH((size_t)9 * 1  * 2  * 512);
    _Float16* p_c2 = allocH((size_t)9 * 2  * 4  * 512);
    _Float16* p_c3 = allocH((size_t)9 * 8  * 16 * 512);
    _Float16* p_c4 = allocH((size_t)9 * 32 * 64 * 512);
    _Float16* p_d4 = allocH((size_t)9 * 16 * 16 * 512);
    _Float16* p_d3 = allocH((size_t)9 * 4  * 4  * 512);
    _Float16* p_d2 = allocH((size_t)9 * 1  * 2  * 512);
    _Float16* p_d1 = allocH((size_t)9 * 1  * 2  * 512);
    float* pb_c1 = allocF(32);   float* pb_c2 = allocF(64);
    float* pb_c3 = allocF(256);  float* pb_c4 = allocF(1024);
    float* pb_d4 = allocF(256);  float* pb_d3 = allocF(64);
    float* pb_d2 = allocF(32);   float* pb_d1 = allocF(32);
    _Float16* zpage = allocH(2048);
    if (used > ws_size) return;

    auto pack = [&](const float* w, _Float16* pk, int Cout, int Cin, int cc, int ct) {
        int total = 9 * cc * ct * 512;
        pack_w_kernel<<<dim3((total + 255) / 256), dim3(256), 0, stream>>>(
            w, pk, Cout, Cin, cc, ct, total);
    };
    auto padb = [&](const float* b, float* o, int Cout, int ct) {
        int n = ct * 16;
        pad_bias_kernel<<<dim3(1), dim3(n), 0, stream>>>(b, o, Cout, n);
    };
    auto conv = [&](const _Float16* in, const _Float16* wp, const float* bp, _Float16* out,
                    const _Float16* add, int H, int W, int inStride, int cc,
                    int outStride, int ct, int addStride, int lrelu) {
        dim3 grid(H * (W / 16), ct / 2);
        conv3x3_wmma_kernel<<<grid, dim3(128), 0, stream>>>(
            in, wp, bp, out, add, zpage, H, W, inStride, cc, outStride, addStride, lrelu);
    };
    auto convL = [&](const _Float16* in, const _Float16* wp, const float* bp, _Float16* out,
                     const _Float16* add, int H, int W, int inStride, int cc,
                     int outStride, int ct, int addStride, int lrelu) {
        dim3 grid(4 * H * (W / 16), ct / 8);   // block = one spatial tile, 8 cout tiles
        conv3x3_wmma_lds_kernel<<<grid, dim3(128), 0, stream>>>(
            in, wp, bp, out, add, zpage, H, W, inStride, cc, outStride, addStride, lrelu);
    };
    auto wtmid = [&](const _Float16* in, _Float16* out, int h2, int w2, int inS, int C, int outS) {
        int total = 4 * h2 * w2 * C;
        wt_mid_kernel<<<dim3((total + 255) / 256), dim3(256), 0, stream>>>(
            in, out, h2, w2, inS, C, outS, total);
    };
    auto iwtcat = [&](const _Float16* in, _Float16* out, int H, int W, int inS, int C,
                      int outS, int chOff) {
        int total = 4 * H * W * C;
        iwt_cat_kernel<<<dim3((total + 255) / 256), dim3(256), 0, stream>>>(
            in, out, H, W, inS, C, outS, chOff, total);
    };

    zero_fill_kernel<<<dim3(4), dim3(256), 0, stream>>>((float*)zpage, 1024);
    pack(w_c1, p_c1, 16,   12,   1,  2);   padb(b_c1, pb_c1, 16,   2);
    pack(w_c2, p_c2, 64,   64,   2,  4);   padb(b_c2, pb_c2, 64,   4);
    pack(w_c3, p_c3, 256,  256,  8,  16);  padb(b_c3, pb_c3, 256,  16);
    pack(w_c4, p_c4, 1024, 1024, 32, 64);  padb(b_c4, pb_c4, 1024, 64);
    pack(w_d4, p_d4, 256,  512,  16, 16);  padb(b_d4, pb_d4, 256,  16);
    pack(w_d3, p_d3, 64,   128,  4,  4);   padb(b_d3, pb_d3, 64,   4);
    pack(w_d2, p_d2, 16,   32,   1,  2);   padb(b_d2, pb_d2, 16,   2);
    pack(w_d1, p_d1, 12,   16,   1,  2);   padb(b_d1, pb_d1, 12,   2);

    // encoder
    wt_in_kernel<<<dim3((4 * 128 * 128 + 255) / 256), dim3(256), 0, stream>>>(x, t_w1);
    conv (t_w1, p_c1, pb_c1, t_iw2, nullptr, 128, 128, 32, 1, 32, 2, 0, 1);       // c1
    wtmid(t_iw2, t_w2, 64, 64, 32, 16, 64);                                       // w2
    conv (t_w2, p_c2, pb_c2, t_iw3, nullptr, 64, 64, 64, 2, 128, 4, 0, 1);        // c2
    wtmid(t_iw3, t_w3, 32, 32, 128, 64, 256);                                     // w3
    convL(t_w3, p_c3, pb_c3, t_iw4, nullptr, 32, 32, 256, 8, 512, 16, 0, 1);      // c3
    wtmid(t_iw4, t_w4, 16, 16, 512, 256, 1024);                                   // w4
    // bottleneck: conv4 x3, last fused with +w4 residual and lrelu
    convL(t_w4, p_c4, pb_c4, t_c4,  nullptr, 16, 16, 1024, 32, 1024, 64, 0,    1); // c4
    convL(t_c4, p_c4, pb_c4, t_c5,  nullptr, 16, 16, 1024, 32, 1024, 64, 0,    1); // c5
    convL(t_c5, p_c4, pb_c4, t_ic4, t_w4,    16, 16, 1024, 32, 1024, 64, 1024, 1); // ic4
    // decoder
    iwtcat(t_ic4, t_iw4, 16, 16, 1024, 256, 512, 256);
    convL(t_iw4, p_d4, pb_d4, t_ic3, nullptr, 32, 32, 512, 16, 256, 16, 0, 1);    // ic3
    iwtcat(t_ic3, t_iw3, 32, 32, 256, 64, 128, 64);
    conv (t_iw3, p_d3, pb_d3, t_ic2, nullptr, 64, 64, 128, 4, 64, 4, 0, 1);       // ic2
    iwtcat(t_ic2, t_iw2, 64, 64, 64, 16, 32, 16);
    conv (t_iw2, p_d2, pb_d2, t_ic1, nullptr, 128, 128, 32, 1, 32, 2, 0, 1);      // ic1
    conv (t_ic1, p_d1, pb_d1, t_iw1, nullptr, 128, 128, 32, 1, 32, 2, 0, 1);      // iw1
    final_kernel<<<dim3((4 * 128 * 128 + 255) / 256), dim3(256), 0, stream>>>(t_iw1, cfin, outp);
}
